// BiDirectionalConcatBlock_73581379715113
// MI455X (gfx1250) — compile-verified, matrix operand order checked by reading, not run on hardware
//
#include <hip/hip_runtime.h>
#include <hip/hip_bf16.h>
#include <math.h>

// ---------------- problem constants (from reference setup) ----------------
#define Bn      2
#define Lseq    2048
#define DM      512            // dm
#define DI      1024           // d_inner
#define RTOT    4096           // Bn * Lseq rows per branch
#define NSTATE  16
#define DTRANK  32
#define NCHUNK  32
#define TCHUNK  64             // Lseq / NCHUNK

typedef __attribute__((ext_vector_type(16))) __bf16       v16bf;
typedef __attribute__((ext_vector_type(8)))  __bf16       v8bf;
typedef __attribute__((ext_vector_type(8)))  float        v8f;
typedef __attribute__((ext_vector_type(4)))  unsigned int v4u;
typedef __attribute__((ext_vector_type(8)))  int          v8i;
typedef __attribute__((ext_vector_type(4)))  int          v4i;

// ---------------- TDM availability probe ----------------
#if defined(__has_builtin)
#if __has_builtin(__builtin_amdgcn_tensor_load_to_lds) && \
    __has_builtin(__builtin_amdgcn_s_wait_tensorcnt)
#define USE_TDM 1
#endif
#endif
#ifndef USE_TDM
#define USE_TDM 0
#endif

// Issue one TDM 2D tile load (f32 elements) into LDS and wait for it.
// width/rows/row_stride in elements. Caller must be a single wave.
__device__ __forceinline__ void tdm_load_2d_f32(void* lds_dst, const float* gsrc,
                                                int width, int rows, int row_stride)
{
#if USE_TDM
  unsigned long long ga = (unsigned long long)(uintptr_t)gsrc;
  unsigned int lds = (unsigned int)(uintptr_t)lds_dst;   // LDS aperture: low 32 bits
  v4u g0;
  g0[0] = 1u;                                            // count=1, user descriptor
  g0[1] = lds;                                           // lds_addr [63:32]
  g0[2] = (unsigned int)(ga & 0xffffffffu);              // global_addr [95:64]
  g0[3] = (unsigned int)((ga >> 32) & 0x1ffffffu)        // global_addr [120:96]
        | 0x80000000u;                                   // type=2 ("image") [127:126]
  v8i g1;
  g1[0] = (int)(2u << 16);                                         // data_size=4B
  g1[1] = (int)(((unsigned)width & 0xffffu) << 16);                // tensor_dim0 lo16
  g1[2] = (int)(((unsigned)width >> 16) |
                (((unsigned)rows & 0xffffu) << 16));               // dim0 hi16 | dim1 lo16
  g1[3] = (int)(((unsigned)rows >> 16) |
                (((unsigned)width & 0xffffu) << 16));              // dim1 hi16 | tile_dim0
  g1[4] = (int)((unsigned)rows & 0xffffu);                         // tile_dim1 (tile_dim2=0)
  g1[5] = (int)(unsigned)row_stride;                               // dim0_stride lo32
  g1[6] = 0;                                                       // stride hi bits
  g1[7] = 0;
  v4i z4 = {0, 0, 0, 0};
#if __clang_major__ >= 23
  v8i z8 = {0, 0, 0, 0, 0, 0, 0, 0};
  __builtin_amdgcn_tensor_load_to_lds(g0, g1, z4, z4, z8, 0);
#else
  __builtin_amdgcn_tensor_load_to_lds(g0, g1, z4, z4, 0);
#endif
  __builtin_amdgcn_s_wait_tensorcnt((short)0);
#else
  (void)lds_dst; (void)gsrc; (void)width; (void)rows; (void)row_stride;
#endif
}

// Build a 16x32 bf16 WMMA fragment slice for this lane from pre-packed bf16.
// p = row_base + kb + 8*half ; elements [0..7] <- p[0..7], [8..15] <- p[16..23]
// (ISA 7.12.2 16-bit A/B layout, wave32).
__device__ __forceinline__ v16bf frag_from_bf16(const __bf16* __restrict__ p) {
  v8bf lo = *(const v8bf*)p;         // 16B load
  v8bf hi = *(const v8bf*)(p + 16);  // 16B load
  v16bf f;
#pragma unroll
  for (int j = 0; j < 8; ++j) { f[j] = lo[j]; f[j + 8] = hi[j]; }
  return f;
}

// C[M,N](f32) = A[M,K](bf16,lda) x W[N,K]^T(bf16,ldw); optional bf16 mirror Cb.
// mode 0: plain. mode 1: softplus(acc + bias[n]).
// Each wave: 16(M) x 64(N) strip, A fragment reused across 4 WMMAs.
__global__ __launch_bounds__(256)
void wmma_gemm_bf16(const __bf16* __restrict__ A, int lda,
                    const __bf16* __restrict__ W, int ldw,
                    float* __restrict__ C, long ldc,
                    __bf16* __restrict__ Cb, long ldcb,
                    int M, int N, int K,
                    const float* __restrict__ bias, int mode)
{
  const int wid  = (int)((blockIdx.x * blockDim.x + threadIdx.x) >> 5);
  const int lane = threadIdx.x & 31;
  const int nStrips = N >> 6;
  const int total   = (M >> 4) * nStrips;
  if (wid >= total) return;
  const int mt   = wid / nStrips;
  const int nt   = wid - mt * nStrips;
  const int row0 = mt << 4;
  const int col0 = nt << 6;
  const int half = lane >> 4;
  const int r    = lane & 15;

  v8f acc[4] = {v8f{}, v8f{}, v8f{}, v8f{}};

  const __bf16* arow = A + (size_t)(row0 + r) * lda + 8 * half;
  const __bf16* w0   = W + (size_t)(col0 +  0 + r) * ldw + 8 * half;
  const __bf16* w1   = W + (size_t)(col0 + 16 + r) * ldw + 8 * half;
  const __bf16* w2   = W + (size_t)(col0 + 32 + r) * ldw + 8 * half;
  const __bf16* w3   = W + (size_t)(col0 + 48 + r) * ldw + 8 * half;

  for (int kb = 0; kb < K; kb += 32) {
    v16bf af = frag_from_bf16(arow + kb);
    v16bf b0 = frag_from_bf16(w0 + kb);
    acc[0] = __builtin_amdgcn_wmma_f32_16x16x32_bf16(false, af, false, b0,
                                                     (short)0, acc[0], false, false);
    v16bf b1 = frag_from_bf16(w1 + kb);
    acc[1] = __builtin_amdgcn_wmma_f32_16x16x32_bf16(false, af, false, b1,
                                                     (short)0, acc[1], false, false);
    v16bf b2 = frag_from_bf16(w2 + kb);
    acc[2] = __builtin_amdgcn_wmma_f32_16x16x32_bf16(false, af, false, b2,
                                                     (short)0, acc[2], false, false);
    v16bf b3 = frag_from_bf16(w3 + kb);
    acc[3] = __builtin_amdgcn_wmma_f32_16x16x32_bf16(false, af, false, b3,
                                                     (short)0, acc[3], false, false);
  }

#pragma unroll
  for (int j = 0; j < 4; ++j) {
    const int n = col0 + 16 * j + r;
    const float bv = (mode == 1) ? bias[n] : 0.f;
#pragma unroll
    for (int v = 0; v < 8; ++v) {
      const int m = row0 + v + 8 * half;     // C/D layout: VGPR v -> M = v + 8*half
      float val = acc[j][v];
      if (mode == 1) {
        val += bv;
        val = (val > 20.f) ? val : log1pf(__expf(val));  // softplus
      }
      C[(size_t)m * ldc + n] = val;
      if (Cb) Cb[(size_t)m * ldcb + n] = (__bf16)val;
    }
  }
}

// f32 -> bf16 pack (weights).
__global__ void pack_bf16(const float* __restrict__ src, __bf16* __restrict__ dst, long n)
{
  long i = (long)blockIdx.x * blockDim.x + threadIdx.x;
  if (i < n) dst[i] = (__bf16)src[i];
}

// Pack branch inputs as bf16: br0 = x[:,:,0:512]; br1 = flip_l(x[:,:,512:1024]).
__global__ void prep_slices(const float* __restrict__ x, __bf16* __restrict__ xsb)
{
  long i = (long)blockIdx.x * blockDim.x + threadIdx.x;
  const long total = 2L * RTOT * DM;
  if (i >= total) return;
  int  c   = (int)(i % DM);
  long t   = i / DM;
  int  row = (int)(t % RTOT);
  int  br  = (int)(t / RTOT);
  int  b   = row / Lseq, l = row % Lseq;
  float v;
  if (br == 0) v = x[((size_t)b * Lseq + l) * (2 * DM) + c];
  else         v = x[((size_t)b * Lseq + (Lseq - 1 - l)) * (2 * DM) + DM + c];
  xsb[i] = (__bf16)v;
}

// Depthwise causal conv (k=4) + bias + SiLU -> xc (f32 for scan) + xcb (bf16 for GEMM).
__global__ void conv_silu(const float* __restrict__ xz,
                          const float* __restrict__ cw1, const float* __restrict__ cb1,
                          const float* __restrict__ cw2, const float* __restrict__ cb2,
                          float* __restrict__ xc, __bf16* __restrict__ xcb)
{
  long i = (long)blockIdx.x * blockDim.x + threadIdx.x;
  const long total = 2L * RTOT * DI;
  if (i >= total) return;
  int  c   = (int)(i % DI);
  long t   = i / DI;
  int  row = (int)(t % RTOT);
  int  br  = (int)(t / RTOT);
  int  l   = row % Lseq;
  const float* cw = (br == 0 ? cw1 : cw2) + c * 4;
  float acc = (br == 0 ? cb1 : cb2)[c];
  const float* xh = xz + ((size_t)(br * RTOT + row)) * 2048 + c;
#pragma unroll
  for (int j = 0; j < 4; ++j) {
    int lo = l - 3 + j;
    if (lo >= 0) acc += cw[j] * xh[(long)(j - 3) * 2048];
  }
  float s = acc / (1.f + __expf(-acc));   // silu
  xc[i]  = s;
  xcb[i] = (__bf16)s;
}

// Pass 1: per-chunk local scan (h_in = 0), emit P = prod(dA) and local h_end.
__global__ __launch_bounds__(1024)
void scan_pass1(const float* __restrict__ delta, const float* __restrict__ xc,
                const float* __restrict__ dbc,   const float* __restrict__ A_log,
                float* __restrict__ chP, float* __restrict__ chH, int br)
{
  const int j = blockIdx.x;             // chunk
  const int b = blockIdx.y;
  const int d = threadIdx.x;            // channel 0..1023
  __shared__ float sBC[TCHUNK * 32];    // per step: [0..15]=B, [16..31]=C
  const int row0 = b * Lseq + j * TCHUNK;
#if USE_TDM
  if (threadIdx.x < 32) {
    tdm_load_2d_f32(sBC, dbc + ((size_t)(br * RTOT + row0)) * 64 + 32,
                    /*width*/32, /*rows*/TCHUNK, /*row_stride*/64);
  }
#else
  for (int i = threadIdx.x; i < TCHUNK * 32; i += blockDim.x) {
    int t = i >> 5, c = i & 31;
    sBC[i] = dbc[((size_t)(br * RTOT + row0 + t)) * 64 + 32 + c];
  }
#endif
  __syncthreads();

  float negA[NSTATE];
#pragma unroll
  for (int s = 0; s < NSTATE; ++s) negA[s] = -__expf(A_log[d * NSTATE + s]);

  float h[NSTATE], P[NSTATE];
#pragma unroll
  for (int s = 0; s < NSTATE; ++s) { h[s] = 0.f; P[s] = 1.f; }

  const float* dp = delta + (size_t)(br * RTOT + row0) * DI + d;
  const float* xp = xc    + (size_t)(br * RTOT + row0) * DI + d;
  for (int t = 0; t < TCHUNK; ++t) {
    float dlt = dp[(size_t)t * DI];
    float xv  = xp[(size_t)t * DI];
    if (t + 1 < TCHUNK) {
      __builtin_prefetch(dp + (size_t)(t + 1) * DI, 0, 1);
      __builtin_prefetch(xp + (size_t)(t + 1) * DI, 0, 1);
    }
    float dx = dlt * xv;
#pragma unroll
    for (int s = 0; s < NSTATE; ++s) {
      float a = __expf(dlt * negA[s]);
      h[s] = h[s] * a + dx * sBC[t * 32 + s];
      P[s] *= a;
    }
  }
  size_t base = ((((size_t)br * NCHUNK + j) * Bn + b) * NSTATE) * DI + d;
#pragma unroll
  for (int s = 0; s < NSTATE; ++s) {
    chP[base + (size_t)s * DI] = P[s];
    chH[base + (size_t)s * DI] = h[s];
  }
}

// Pass 2: tiny sequential inter-chunk carry:  hc_j = P_j * hc_{j-1} + H_j.
__global__ __launch_bounds__(1024)
void scan_carry(const float* __restrict__ chP, const float* __restrict__ chH,
                float* __restrict__ hin, int br)
{
  const int b = blockIdx.x;
  const int d = threadIdx.x;
  float hc[NSTATE];
#pragma unroll
  for (int s = 0; s < NSTATE; ++s) hc[s] = 0.f;
  for (int j = 0; j < NCHUNK; ++j) {
    size_t base = ((((size_t)br * NCHUNK + j) * Bn + b) * NSTATE) * DI + d;
#pragma unroll
    for (int s = 0; s < NSTATE; ++s) {
      size_t idx = base + (size_t)s * DI;
      hin[idx] = hc[s];
      hc[s] = chP[idx] * hc[s] + chH[idx];
    }
  }
}

// Pass 3: replay chunk with correct h_in; y = (h.C + xc*D) * silu(z), stored bf16.
__global__ __launch_bounds__(1024)
void scan_pass3(const float* __restrict__ delta, const float* __restrict__ xc,
                const float* __restrict__ dbc,   const float* __restrict__ A_log,
                const float* __restrict__ Dp,    const float* __restrict__ hin,
                const float* __restrict__ xz,    __bf16* __restrict__ yb, int br)
{
  const int j = blockIdx.x;
  const int b = blockIdx.y;
  const int d = threadIdx.x;
  __shared__ float sBC[TCHUNK * 32];
  const int row0 = b * Lseq + j * TCHUNK;
#if USE_TDM
  if (threadIdx.x < 32) {
    tdm_load_2d_f32(sBC, dbc + ((size_t)(br * RTOT + row0)) * 64 + 32,
                    32, TCHUNK, 64);
  }
#else
  for (int i = threadIdx.x; i < TCHUNK * 32; i += blockDim.x) {
    int t = i >> 5, c = i & 31;
    sBC[i] = dbc[((size_t)(br * RTOT + row0 + t)) * 64 + 32 + c];
  }
#endif
  __syncthreads();

  float negA[NSTATE];
#pragma unroll
  for (int s = 0; s < NSTATE; ++s) negA[s] = -__expf(A_log[d * NSTATE + s]);

  float h[NSTATE];
  size_t hbase = ((((size_t)br * NCHUNK + j) * Bn + b) * NSTATE) * DI + d;
#pragma unroll
  for (int s = 0; s < NSTATE; ++s) h[s] = hin[hbase + (size_t)s * DI];

  const float Dc = Dp[d];
  const float* dp = delta + (size_t)(br * RTOT + row0) * DI + d;
  const float* xp = xc    + (size_t)(br * RTOT + row0) * DI + d;
  const float* zp = xz    + (size_t)(br * RTOT + row0) * 2048 + 1024 + d;
  __bf16*      yp = yb    + (size_t)(br * RTOT + row0) * DI + d;
  for (int t = 0; t < TCHUNK; ++t) {
    float dlt = dp[(size_t)t * DI];
    float xv  = xp[(size_t)t * DI];
    float zv  = zp[(size_t)t * 2048];
    if (t + 1 < TCHUNK) {
      __builtin_prefetch(dp + (size_t)(t + 1) * DI, 0, 1);
      __builtin_prefetch(xp + (size_t)(t + 1) * DI, 0, 1);
      __builtin_prefetch(zp + (size_t)(t + 1) * 2048, 0, 1);
    }
    float dx = dlt * xv;
    float yt = 0.f;
#pragma unroll
    for (int s = 0; s < NSTATE; ++s) {
      float a = __expf(dlt * negA[s]);
      h[s] = h[s] * a + dx * sBC[t * 32 + s];
      yt  += h[s] * sBC[t * 32 + 16 + s];
    }
    float sz = zv / (1.f + __expf(-zv));
    yp[(size_t)t * DI] = (__bf16)((yt + xv * Dc) * sz);
  }
}

__global__ void copy_residual(const float4* __restrict__ x, float4* __restrict__ out, long n4)
{
  long i = (long)blockIdx.x * blockDim.x + threadIdx.x;
  if (i < n4) out[i] = x[i];
}

extern "C" void kernel_launch(void* const* d_in, const int* in_sizes, int n_in,
                              void* d_out, int out_size, void* d_ws, size_t ws_size,
                              hipStream_t stream)
{
  (void)in_sizes; (void)n_in; (void)out_size; (void)ws_size;
  const float* x          = (const float*)d_in[0];
  const float* in_w[2]    = {(const float*)d_in[1],  (const float*)d_in[10]};
  const float* conv_w[2]  = {(const float*)d_in[2],  (const float*)d_in[11]};
  const float* conv_b[2]  = {(const float*)d_in[3],  (const float*)d_in[12]};
  const float* xproj_w[2] = {(const float*)d_in[4],  (const float*)d_in[13]};
  const float* dt_w[2]    = {(const float*)d_in[5],  (const float*)d_in[14]};
  const float* dt_b[2]    = {(const float*)d_in[6],  (const float*)d_in[15]};
  const float* A_log[2]   = {(const float*)d_in[7],  (const float*)d_in[16]};
  const float* Dp[2]      = {(const float*)d_in[8],  (const float*)d_in[17]};
  const float* out_w[2]   = {(const float*)d_in[9],  (const float*)d_in[18]};

  // ---------------- workspace carve (256B aligned chunks) ----------------
  char* p = (char*)d_ws;
  auto alloc = [&](size_t bytes) -> void* {
    void* q = (void*)p; p += (bytes + 255) & ~(size_t)255; return q;
  };
  float*  xz    = (float*) alloc(2ULL * RTOT * 2048 * 4);          // xh | z
  float*  xc    = (float*) alloc(2ULL * RTOT * DI * 4);
  float*  delta = (float*) alloc(2ULL * RTOT * DI * 4);
  float*  dbc   = (float*) alloc(2ULL * RTOT * 64 * 4);
  float*  chP   = (float*) alloc(2ULL * NCHUNK * Bn * NSTATE * DI * 4);
  float*  chH   = (float*) alloc(2ULL * NCHUNK * Bn * NSTATE * DI * 4);
  float*  hin   = (float*) alloc(2ULL * NCHUNK * Bn * NSTATE * DI * 4);
  __bf16* xsb   = (__bf16*)alloc(2ULL * RTOT * DM * 2);
  __bf16* xcb   = (__bf16*)alloc(2ULL * RTOT * DI * 2);
  __bf16* dbcb  = (__bf16*)alloc(2ULL * RTOT * 64 * 2);
  __bf16* ybb   = (__bf16*)alloc(2ULL * RTOT * DI * 2);
  __bf16* inwb  = (__bf16*)alloc(2ULL * 2048 * DM * 2);
  __bf16* xpwb  = (__bf16*)alloc(2ULL * 64 * DI * 2);
  __bf16* dtwb  = (__bf16*)alloc(2ULL * DI * DTRANK * 2);
  __bf16* outwb = (__bf16*)alloc(2ULL * DM * DI * 2);

  float* hidden = (float*)d_out;                      // [B][L][1024]
  float* resid  = hidden + (size_t)RTOT * 1024;       // [B][L][1024]

  auto pack = [&](const float* src, __bf16* dst, long n) {
    pack_bf16<<<(int)((n + 255) / 256), 256, 0, stream>>>(src, dst, n);
  };
  auto launch_gemm = [&](const __bf16* A, int lda, const __bf16* W, int ldw,
                         float* C, long ldc, __bf16* Cb, long ldcb,
                         int M, int N, int K, const float* bias, int mode) {
    int waves  = (M / 16) * (N / 64);
    int blocks = (waves * 32 + 255) / 256;
    wmma_gemm_bf16<<<blocks, 256, 0, stream>>>(A, lda, W, ldw, C, ldc, Cb, ldcb,
                                               M, N, K, bias, mode);
  };

  // 0) pack weights to bf16 + pack branch inputs (branch 1 flipped along L)
  for (int br = 0; br < 2; ++br) {
    pack(in_w[br],    inwb  + (size_t)br * 2048 * DM,  2048L * DM);
    pack(xproj_w[br], xpwb  + (size_t)br * 64 * DI,    64L * DI);
    pack(dt_w[br],    dtwb  + (size_t)br * DI * DTRANK, (long)DI * DTRANK);
    pack(out_w[br],   outwb + (size_t)br * DM * DI,    (long)DM * DI);
  }
  { long n = 2L * RTOT * DM;
    prep_slices<<<(int)((n + 255) / 256), 256, 0, stream>>>(x, xsb); }

  // 1) in_proj: xz = xs @ in_w^T   (M=4096, N=2048, K=512)
  for (int br = 0; br < 2; ++br)
    launch_gemm(xsb + (size_t)br * RTOT * DM, DM, inwb + (size_t)br * 2048 * DM, DM,
                xz + (size_t)br * RTOT * 2048, 2048, nullptr, 0,
                RTOT, 2048, DM, nullptr, 0);

  // 2) depthwise causal conv + SiLU -> xc (f32) + xcb (bf16)
  { long n = 2L * RTOT * DI;
    conv_silu<<<(int)((n + 255) / 256), 256, 0, stream>>>(
        xz, conv_w[0], conv_b[0], conv_w[1], conv_b[1], xc, xcb); }

  // 3) x_proj: dbc = xc @ xproj_w^T   (N=64, K=1024); bf16 mirror for dt GEMM
  for (int br = 0; br < 2; ++br)
    launch_gemm(xcb + (size_t)br * RTOT * DI, DI, xpwb + (size_t)br * 64 * DI, DI,
                dbc + (size_t)br * RTOT * 64, 64,
                dbcb + (size_t)br * RTOT * 64, 64,
                RTOT, 64, DI, nullptr, 0);

  // 4) delta = softplus(dt @ dt_w^T + dt_b)   (N=1024, K=32)
  for (int br = 0; br < 2; ++br)
    launch_gemm(dbcb + (size_t)br * RTOT * 64, 64, dtwb + (size_t)br * DI * DTRANK, DTRANK,
                delta + (size_t)br * RTOT * DI, DI, nullptr, 0,
                RTOT, DI, DTRANK, dt_b[br], 1);

  // 5) chunked selective scan (3 passes; B/C tile via TDM when available)
  for (int br = 0; br < 2; ++br)
    scan_pass1<<<dim3(NCHUNK, Bn), 1024, 0, stream>>>(delta, xc, dbc, A_log[br], chP, chH, br);
  for (int br = 0; br < 2; ++br)
    scan_carry<<<Bn, 1024, 0, stream>>>(chP, chH, hin, br);
  for (int br = 0; br < 2; ++br)
    scan_pass3<<<dim3(NCHUNK, Bn), 1024, 0, stream>>>(delta, xc, dbc, A_log[br], Dp[br],
                                                      hin, xz, ybb, br);

  // 6) out_proj into concatenated hidden halves  (N=512, K=1024, ldc=1024)
  for (int br = 0; br < 2; ++br)
    launch_gemm(ybb + (size_t)br * RTOT * DI, DI, outwb + (size_t)br * DM * DI, DI,
                hidden + (size_t)br * DM, 1024, nullptr, 0,
                RTOT, DM, DI, nullptr, 0);

  // 7) residual passthrough
  { long n4 = (long)RTOT * 1024 / 4;
    copy_residual<<<(int)((n4 + 255) / 256), 256, 0, stream>>>(
        (const float4*)x, (float4*)resid, n4); }
}
